// FixedStrideMeanChunker_63788854280487
// MI455X (gfx1250) — matrix operand order repaired and checked by reading, not compiled, not run on hardware
//
#include <hip/hip_runtime.h>
#include <hip/hip_bf16.h>

// ---------------- types for WMMA ----------------
typedef __attribute__((ext_vector_type(16))) __bf16        v16bf;
typedef __attribute__((ext_vector_type(8)))  float         v8f;
typedef __attribute__((ext_vector_type(4)))  unsigned int  u32x4;

union Frag16 {           // 16 bf16 elements == two 128-bit chunks
    v16bf bf;
    u32x4 q[2];
};

static __device__ __forceinline__ unsigned short f32_to_bf16(float f) {
    unsigned int u = __float_as_uint(f);
    u += 0x7FFFu + ((u >> 16) & 1u);   // round to nearest even
    return (unsigned short)(u >> 16);
}

// Problem constants (from setup_inputs): B=16, T=4096, D=512, stride=8
#define BATCH   16
#define TLEN    4096
#define DDIM    512
#define STRIDE  8
#define NWIN    (TLEN / STRIDE)        // 512
#define MROWS   (BATCH * NWIN)         // 8192
#define KSTEPS  (DDIM / 32)            // 16

// ---------------------------------------------------------------
// Kernel 1: masked window mean  -> bf16 mean matrix + active flags
// one 256-thread block per window; each thread owns 2 columns
// ---------------------------------------------------------------
__global__ void mean_kernel(const float* __restrict__ frames,
                            unsigned short* __restrict__ mean_bf,
                            float* __restrict__ flags) {
    const int win = blockIdx.x;                 // 0..8191
    const int b   = win >> 9;                   // /512
    const int w   = win & (NWIN - 1);
    const int tid = threadIdx.x;                // 0..255
    const int c0  = tid * 2;

    const float* base = frames + ((size_t)(b * TLEN + w * STRIDE)) * DDIM;

    float x0[STRIDE], x1[STRIDE], pabs[STRIDE];
#pragma unroll
    for (int r = 0; r < STRIDE; ++r) {
        const float2 v = *(const float2*)(base + (size_t)r * DDIM + c0);
        x0[r] = v.x; x1[r] = v.y;
        pabs[r] = fabsf(v.x) + fabsf(v.y);
    }

    __shared__ float rowsum[STRIDE];
    if (tid < STRIDE) rowsum[tid] = 0.0f;
    __syncthreads();

    // wave32 reduce, then one ds_add_f32 per wave per row
#pragma unroll
    for (int r = 0; r < STRIDE; ++r) {
        float v = pabs[r];
#pragma unroll
        for (int off = 16; off > 0; off >>= 1) v += __shfl_xor(v, off, 32);
        if ((tid & 31) == 0) atomicAdd(&rowsum[r], v);
    }
    __syncthreads();

    float cnt = 0.0f, s0 = 0.0f, s1 = 0.0f;
#pragma unroll
    for (int r = 0; r < STRIDE; ++r) {
        const float valid = (rowsum[r] != 0.0f) ? 1.0f : 0.0f;
        cnt += valid;
        s0  += valid * x0[r];
        s1  += valid * x1[r];
    }
    const float denom = fmaxf(cnt, 1.0f);
    const unsigned int lo = f32_to_bf16(s0 / denom);
    const unsigned int hi = f32_to_bf16(s1 / denom);
    *(unsigned int*)(mean_bf + (size_t)win * DDIM + c0) = lo | (hi << 16);

    if (tid == 0) flags[win] = (cnt > 0.0f) ? 1.0f : 0.0f;
}

// ---------------------------------------------------------------
// Kernel 2/3: GEMM  C[M,N] = act(A[M,K] * W[K,N] + bias) (bf16 WMMA)
//   FIRST = true : ReLU, store bf16 to Hout
//   FIRST = false: multiply by window-active flag, store f32 to Fout
// Workgroup: 256 threads = 8 waves, tile 128(M) x 64(N), K step 32.
// Double-buffered LDS + register prefetch: one barrier per K step.
// ---------------------------------------------------------------
template <bool FIRST>
__global__ __launch_bounds__(256)
void mlp_gemm(const unsigned short* __restrict__ Abf,   // [M,K] bf16
              const float* __restrict__ W,              // [K,N] f32
              const float* __restrict__ bias,           // [N]
              const float* __restrict__ flags,          // [M]
              unsigned short* __restrict__ Hout,        // [M,N] bf16 (FIRST)
              float* __restrict__ Fout) {               // [M,N] f32  (!FIRST)
    const int K = DDIM, N = DDIM;
    const int m0 = blockIdx.x * 128;
    const int n0 = blockIdx.y * 64;

    const int tid  = threadIdx.x;
    const int wave = tid >> 5;          // 0..7 -> M strips
    const int lane = tid & 31;
    const int half = lane >> 4;         // 0/1
    const int lrow = lane & 15;
    const int koffA = half * 16;        // A: K {koffA..koffA+15} contiguous
    const int koffB = half * 16;        // B: K {koffB..koffB+15} contiguous

    // staging thread mapping (same for all tiles)
    const int klocal = tid & 31;        // B: k within tile
    const int ng     = (tid >> 5) * 8;  // B: n group of 8

    __shared__ unsigned short Alds[2][128 * 32];   // 2 x 8 KB, row-major [m][k]
    __shared__ unsigned short Blds[2][64 * 32];    // 2 x 4 KB, N-major   [n][k]

    // register prefetch staging
    u32x4  aPre[2];
    float4 bPre0, bPre1;

    auto load_tile = [&](int kk) {
#pragma unroll
        for (int i = 0; i < 2; ++i) {
            const int ci  = tid + i * 256;      // 0..511
            const int row = ci >> 2;
            const int k8  = (ci & 3) << 3;
            aPre[i] = *(const u32x4*)(Abf + (size_t)(m0 + row) * K + kk + k8);
        }
        const float* src = W + (size_t)(kk + klocal) * N + n0 + ng;
        bPre0 = *(const float4*)src;
        bPre1 = *(const float4*)(src + 4);
    };

    auto store_tile = [&](int buf) {
#pragma unroll
        for (int i = 0; i < 2; ++i) {
            const int ci  = tid + i * 256;
            const int row = ci >> 2;
            const int k8  = (ci & 3) << 3;
            *(u32x4*)&Alds[buf][row * 32 + k8] = aPre[i];
        }
        unsigned short* bl = &Blds[buf][0];
        bl[(ng + 0) * 32 + klocal] = f32_to_bf16(bPre0.x);
        bl[(ng + 1) * 32 + klocal] = f32_to_bf16(bPre0.y);
        bl[(ng + 2) * 32 + klocal] = f32_to_bf16(bPre0.z);
        bl[(ng + 3) * 32 + klocal] = f32_to_bf16(bPre0.w);
        bl[(ng + 4) * 32 + klocal] = f32_to_bf16(bPre1.x);
        bl[(ng + 5) * 32 + klocal] = f32_to_bf16(bPre1.y);
        bl[(ng + 6) * 32 + klocal] = f32_to_bf16(bPre1.z);
        bl[(ng + 7) * 32 + klocal] = f32_to_bf16(bPre1.w);
    };

    v8f acc[4];
#pragma unroll
    for (int j = 0; j < 4; ++j) acc[j] = (v8f){0.f,0.f,0.f,0.f,0.f,0.f,0.f,0.f};

    // prologue: stage tile 0
    load_tile(0);
    store_tile(0);
    __syncthreads();

    const int arow = wave * 16 + lrow;

    for (int it = 0; it < KSTEPS; ++it) {
        const int buf = it & 1;

        // issue next tile's global loads early (hide latency under WMMA)
        if (it + 1 < KSTEPS) load_tile((it + 1) * 32);

        // ---- load ALL fragments first, then back-to-back WMMAs
        Frag16 a;
        a.q[0] = *(const u32x4*)&Alds[buf][arow * 32 + koffA];
        a.q[1] = *(const u32x4*)&Alds[buf][arow * 32 + koffA + 8];

        Frag16 bfr[4];
#pragma unroll
        for (int j = 0; j < 4; ++j) {
            const int bn = j * 16 + lrow;
            bfr[j].q[0] = *(const u32x4*)&Blds[buf][bn * 32 + koffB];
            bfr[j].q[1] = *(const u32x4*)&Blds[buf][bn * 32 + koffB + 8];
        }
#pragma unroll
        for (int j = 0; j < 4; ++j) {
            acc[j] = __builtin_amdgcn_wmma_f32_16x16x32_bf16(
                false, a.bf, false, bfr[j].bf, (short)0, acc[j], false, false);
        }

        // store prefetched tile into the other buffer; one barrier per step
        if (it + 1 < KSTEPS) store_tile(buf ^ 1);
        __syncthreads();
    }

    // ---- epilogue (C layout: lane col = lrow; VGPR v -> row v or v+8)
#pragma unroll
    for (int j = 0; j < 4; ++j) {
        const int col = n0 + j * 16 + lrow;
        const float bb = bias[col];
#pragma unroll
        for (int v = 0; v < 8; ++v) {
            const int row = m0 + wave * 16 + (half ? v + 8 : v);
            float val = acc[j][v] + bb;
            if (FIRST) {
                val = fmaxf(val, 0.0f);
                Hout[(size_t)row * N + col] = f32_to_bf16(val);
            } else {
                val *= flags[row];
                Fout[(size_t)row * N + col] = val;
            }
        }
    }
}

// ---------------------------------------------------------------
extern "C" void kernel_launch(void* const* d_in, const int* in_sizes, int n_in,
                              void* d_out, int out_size, void* d_ws, size_t ws_size,
                              hipStream_t stream) {
    const float* frames = (const float*)d_in[0];
    const float* W1     = (const float*)d_in[1];
    const float* b1     = (const float*)d_in[2];
    const float* W2     = (const float*)d_in[3];
    const float* b2     = (const float*)d_in[4];
    float* out = (float*)d_out;

    // workspace layout
    unsigned short* mean_bf = (unsigned short*)d_ws;                       // 8 MB
    unsigned short* h_bf    = mean_bf + (size_t)MROWS * DDIM;              // 8 MB
    float*          flags   = (float*)(h_bf + (size_t)MROWS * DDIM);       // 32 KB

    // 1) masked window mean -> bf16 + flags
    mean_kernel<<<MROWS, 256, 0, stream>>>(frames, mean_bf, flags);

    // 2) H = relu(mean * W1 + b1)   (bf16 WMMA, bf16 out)
    dim3 grid(MROWS / 128, DDIM / 64);
    mlp_gemm<true><<<grid, 256, 0, stream>>>(mean_bf, W1, b1, flags, h_bf, nullptr);

    // 3) OUT = (H * W2 + b2) * active  (bf16 WMMA, f32 out)
    mlp_gemm<false><<<grid, 256, 0, stream>>>(h_bf, W2, b2, flags, nullptr, out);
}
